// AttentionFlowLayer_48498770707203
// MI455X (gfx1250) — matrix operand order repaired and checked by reading, not compiled, not run on hardware
//
#include <hip/hip_runtime.h>
#include <hip/hip_bf16.h>

// ---------------- problem constants ----------------
#define BB   16
#define TP   4096
#define TQ   512
#define DD   256
#define ROWS 64      // i-rows per workgroup in attn kernel
#define JCH  64      // j-chunk for flash loop
#define GW   1024    // G row width = 4*DD

typedef __attribute__((ext_vector_type(16))) __bf16 v16bf;
typedef __attribute__((ext_vector_type(8)))  float  v8f;
typedef __attribute__((ext_vector_type(4)))  unsigned int u32x4;
typedef __attribute__((ext_vector_type(8)))  int i32x8;
typedef __attribute__((ext_vector_type(4)))  int i32x4;

union ABFrag { v16bf v; uint4 q[2]; };

__device__ __forceinline__ unsigned short f2bf(float f) {
  unsigned int u = __float_as_uint(f);
  unsigned int r = u + 0x7FFFu + ((u >> 16) & 1u);   // round-to-nearest-even
  return (unsigned short)(r >> 16);
}

// Low 32 bits of a flat shared-memory address = LDS byte offset (ISA 10.2).
__device__ __forceinline__ unsigned lds_off(const void* p) {
  return (unsigned)(unsigned long long)p;
}

// --- Tensor DMA descriptor helpers (CDNA5 ISA ch.8) ---
__device__ __forceinline__ u32x4 tdm_g0(unsigned lds_addr, const void* gptr) {
  unsigned long long ga = (unsigned long long)gptr;
  u32x4 g;
  g[0] = 1u;                                        // count=1, user mode
  g[1] = lds_addr;                                  // lds_addr [63:32]
  g[2] = (unsigned)(ga & 0xFFFFFFFFu);              // global_addr lo
  g[3] = (unsigned)((ga >> 32) & 0x01FFFFFFu)       // global_addr [56:32]
         | 0x80000000u;                             // type=2 (bits 127:126)
  return g;
}

// 1-D contiguous byte copy of `nbytes` (<= 65535)
__device__ __forceinline__ i32x8 tdm_g1_1d(unsigned nbytes) {
  i32x8 g;
  g[0] = 0;                               // mask=0, data_size=0 (1B), no flags
  g[1] = (int)(nbytes << 16);             // tensor_dim0[15:0]
  g[2] = (int)(1u << 16);                 // tensor_dim0 hi=0, tensor_dim1=1
  g[3] = (int)(nbytes << 16);             // tensor_dim1 hi=0, tile_dim0
  g[4] = 0;                               // tile_dim1=0 (unused), tile_dim2=0
  g[5] = (int)nbytes;                     // tensor_dim0_stride lo
  g[6] = 0;                               // stride0 hi, tensor_dim1_stride lo
  g[7] = 0;
  return g;
}

// 2-D copy: elem=2B, tile 64(x) x 256(y), tensor_dim0=512, row stride 512 elems
__device__ __forceinline__ i32x8 tdm_g1_ut(void) {
  i32x8 g;
  g[0] = (int)(1u << 16);                 // data_size=1 -> 2 bytes
  g[1] = (int)(512u << 16);               // tensor_dim0 = 512
  g[2] = (int)(256u << 16);               // tensor_dim1 = 256
  g[3] = (int)(64u << 16);                // tile_dim0 = 64
  g[4] = 256;                             // tile_dim1 = 256, tile_dim2 = 0
  g[5] = 512;                             // tensor_dim0_stride = 512
  g[6] = 0;
  g[7] = 0;
  return g;
}

__device__ __forceinline__ void tdm_load(u32x4 g0, i32x8 g1) {
  i32x4 z4; z4[0] = 0; z4[1] = 0; z4[2] = 0; z4[3] = 0;
#if defined(__clang_major__) && (__clang_major__ >= 23)
  i32x8 z8; for (int k = 0; k < 8; ++k) z8[k] = 0;
  __builtin_amdgcn_tensor_load_to_lds(g0, g1, z4, z4, z8, 0);
#else
  __builtin_amdgcn_tensor_load_to_lds(g0, g1, z4, z4, 0);
#endif
}

// ---------------------------------------------------------------------------
// Kernel 1: mask U, produce bf16 U (row-major [b][j][c]), bf16 U^T
// ([b][c][j], j contiguous) and cU[b][j] = Um . w_u
// ---------------------------------------------------------------------------
__global__ __launch_bounds__(256)
void prep_u_kernel(const float* __restrict__ U, const float* __restrict__ w,
                   const int* __restrict__ mask_q,
                   unsigned short* __restrict__ Ubf,
                   unsigned short* __restrict__ Utbf,
                   float* __restrict__ cU)
{
  int idx = blockIdx.x * 256 + threadIdx.x;   // 0 .. BB*TQ-1
  int b = idx / TQ;
  int j = idx % TQ;
  float qm = (float)mask_q[idx];
  const float* ur = U + (size_t)idx * DD;
  float acc = 0.f;
  for (int c = 0; c < DD; ++c) {
    float um = ur[c] * qm;
    acc += um * w[DD + c];                    // w_u = w[256..511]
    unsigned short bv = f2bf(um);
    Ubf[(size_t)idx * DD + c] = bv;
    Utbf[((size_t)b * DD + c) * TQ + j] = bv; // coalesced across threads (j)
  }
  cU[idx] = acc;
}

// ---------------------------------------------------------------------------
// Kernel 2: flash attention over j for a 64-row i tile.
//   TDM double-buffers the U / U^T j-chunks into LDS while WMMA computes.
// ---------------------------------------------------------------------------
__global__ __launch_bounds__(256)
void attn_kernel(const float* __restrict__ H, const float* __restrict__ w,
                 const int* __restrict__ mask_p, const int* __restrict__ mask_q,
                 const unsigned short* __restrict__ Ubf,
                 const unsigned short* __restrict__ Utbf,
                 const float* __restrict__ cU,
                 float* __restrict__ Mrow,
                 float* __restrict__ G)
{
  __shared__ unsigned short Qs[ROWS * DD];        // 32 KB  bf16 Q tile
  __shared__ float          Sc[ROWS * JCH];       // 16 KB  S chunk
  __shared__ unsigned short Ps[ROWS * JCH];       //  8 KB  bf16 probabilities
  __shared__ float          red[ROWS * 4];        //  1 KB  row partials
  __shared__ float mrow[ROWS], lrow[ROWS], srow[ROWS], rH[ROWS];
  __shared__ unsigned short Ustage[2][JCH * DD];  // 64 KB  U[j][c] chunk, x2
  __shared__ unsigned short Utstage[2][DD * JCH]; // 64 KB  U^T[c][j] chunk, x2

  const int b   = blockIdx.y;
  const int i0  = blockIdx.x * ROWS;
  const int tid = threadIdx.x;
  const int lane = tid & 31;
  const int wv   = tid >> 5;        // 8 waves
  const int r    = lane & 15;
  const int half = lane >> 4;

  const unsigned short* Ub  = Ubf  + (size_t)b * TQ * DD;   // [j][c]
  const unsigned short* Utb = Utbf + (size_t)b * DD * TQ;   // [c][j]

  // Kick off TDM for the first j-chunk (wave 0 only; TDM ignores EXEC).
  if (wv == 0) {
    tdm_load(tdm_g0(lds_off(&Ustage[0][0]),  Ub),        tdm_g1_1d(JCH * DD * 2));
    tdm_load(tdm_g0(lds_off(&Utstage[0][0]), Utb),       tdm_g1_ut());
  }

  if (tid < ROWS) { mrow[tid] = -3.0e38f; lrow[tid] = 0.f; rH[tid] = 0.f; }
  __syncthreads();

  // ---- Phase 1: load H tile -> Q (bf16, LDS) + rH partial dot ----
  {
    const int row = tid >> 2;
    const int seg = tid & 3;
    const float pm = (float)mask_p[b * TP + i0 + row];
    const float* hrow = H + ((size_t)b * TP + i0 + row) * DD + seg * 64;
    float prh = 0.f;
    for (int c = 0; c < 64; c += 4) {
      const int cg = seg * 64 + c;
      float4 hv = *(const float4*)(hrow + c);
      hv.x *= pm; hv.y *= pm; hv.z *= pm; hv.w *= pm;
      prh += hv.x * w[cg] + hv.y * w[cg + 1] + hv.z * w[cg + 2] + hv.w * w[cg + 3];
      Qs[row * DD + cg + 0] = f2bf(hv.x * w[2 * DD + cg + 0]);   // w_hu = w[512..]
      Qs[row * DD + cg + 1] = f2bf(hv.y * w[2 * DD + cg + 1]);
      Qs[row * DD + cg + 2] = f2bf(hv.z * w[2 * DD + cg + 2]);
      Qs[row * DD + cg + 3] = f2bf(hv.w * w[2 * DD + cg + 3]);
    }
    atomicAdd(&rH[row], prh);
  }

  // wave -> (row block, column half) mapping
  const int rb = wv >> 1;   // 0..3  (16 i-rows each)
  const int ch = wv & 1;    // 0..1  (128 channels each)

  v8f Oa[8];                // 8 output tiles (16x16) per wave
  for (int t = 0; t < 8; ++t)
    for (int v = 0; v < 8; ++v) Oa[t][v] = 0.f;

  for (int j0 = 0, cc = 0; j0 < TQ; j0 += JCH, ++cc) {
    const int bi = cc & 1;

    // Issue TDM for the next chunk, then wait for the current one (in-order).
    if (wv == 0) {
      if (j0 + JCH < TQ) {
        tdm_load(tdm_g0(lds_off(&Ustage[bi ^ 1][0]),
                        Ub + (size_t)(j0 + JCH) * DD), tdm_g1_1d(JCH * DD * 2));
        tdm_load(tdm_g0(lds_off(&Utstage[bi ^ 1][0]),
                        Utb + (j0 + JCH)),             tdm_g1_ut());
        __builtin_amdgcn_s_wait_tensorcnt(2);
      } else {
        __builtin_amdgcn_s_wait_tensorcnt(0);
      }
    }
    __syncthreads();   // staged chunk visible to all waves

    const unsigned short* Us  = &Ustage[bi][0];
    const unsigned short* Uts = &Utstage[bi][0];

    // ---- Phase A: S chunk via WMMA (2 tiles per wave, k = 256 in 8 steps) --
    for (int tcb = 0; tcb < 2; ++tcb) {
      const int cb = ch * 2 + tcb;
      const int jcol = j0 + cb * 16 + r;
      v8f s;
      for (int v = 0; v < 8; ++v) s[v] = 0.f;
      for (int k = 0; k < 8; ++k) {
        ABFrag af, bfrag;
        const unsigned short* qp = &Qs[(rb * 16 + r) * DD + k * 32 + 8 * half];
        af.q[0] = *(const uint4*)qp;
        af.q[1] = *(const uint4*)(qp + 16);
        const unsigned short* bp = Us + (cb * 16 + r) * DD + k * 32 + 8 * half;
        bfrag.q[0] = *(const uint4*)bp;
        bfrag.q[1] = *(const uint4*)(bp + 16);
        s = __builtin_amdgcn_wmma_f32_16x16x32_bf16(
                false, af.v, false, bfrag.v, (short)0, s, false, false);
      }
      const float cu = cU[b * TQ + jcol];
      for (int v = 0; v < 8; ++v)
        Sc[(rb * 16 + 8 * half + v) * JCH + cb * 16 + r] = s[v] + cu;
    }
    __syncthreads();

    // ---- Phase B: online softmax bookkeeping (thread = row quarter) -------
    {
      const int row = tid >> 2, seg = tid & 3;
      float lm = -3.0e38f;
      for (int c = 0; c < 16; ++c)
        lm = fmaxf(lm, Sc[row * JCH + seg * 16 + c]);
      red[row * 4 + seg] = lm;
    }
    __syncthreads();
    if (tid < ROWS) {
      float cm = fmaxf(fmaxf(red[tid * 4 + 0], red[tid * 4 + 1]),
                       fmaxf(red[tid * 4 + 2], red[tid * 4 + 3]));
      float mo = mrow[tid];
      float mn = fmaxf(mo, cm);
      srow[tid] = __expf(mo - mn);
      mrow[tid] = mn;
    }
    __syncthreads();
    {
      const int row = tid >> 2, seg = tid & 3;
      const float mn = mrow[row];
      float ps = 0.f;
      for (int c = 0; c < 16; ++c) {
        const int j = j0 + seg * 16 + c;
        const float qm = (float)mask_q[b * TQ + j];
        const float p = qm * __expf(Sc[row * JCH + seg * 16 + c] - mn);
        Ps[row * JCH + seg * 16 + c] = f2bf(p);
        ps += p;
      }
      red[row * 4 + seg] = ps;
    }
    __syncthreads();
    if (tid < ROWS) {
      lrow[tid] = lrow[tid] * srow[tid] +
                  (red[tid * 4 + 0] + red[tid * 4 + 1] +
                   red[tid * 4 + 2] + red[tid * 4 + 3]);
    }
    __syncthreads();

    // ---- Phase C: rescale O and accumulate P @ V via WMMA -----------------
    {
      float scl[8];
      for (int v = 0; v < 8; ++v) scl[v] = srow[rb * 16 + 8 * half + v];
      for (int t = 0; t < 8; ++t)
        for (int v = 0; v < 8; ++v) Oa[t][v] *= scl[v];
    }
    for (int tt = 0; tt < 8; ++tt) {
      const int c0 = (ch * 8 + tt) * 16;
      for (int kk = 0; kk < 2; ++kk) {
        ABFrag af, bfrag;
        const unsigned short* pp = &Ps[(rb * 16 + r) * JCH + kk * 32 + 8 * half];
        af.q[0] = *(const uint4*)pp;
        af.q[1] = *(const uint4*)(pp + 16);
        const unsigned short* bp = Uts + (c0 + r) * JCH + kk * 32 + 8 * half;
        bfrag.q[0] = *(const uint4*)bp;
        bfrag.q[1] = *(const uint4*)(bp + 16);
        Oa[tt] = __builtin_amdgcn_wmma_f32_16x16x32_bf16(
                     false, af.v, false, bfrag.v, (short)0, Oa[tt], false, false);
      }
    }
    __syncthreads();   // Sc/Ps/stage buffers reused next chunk
  }

  // ---- Epilogue: M row stats, U_att slice, Hm slice ----
  if (tid < ROWS)
    Mrow[b * TP + i0 + tid] = rH[tid] + mrow[tid];

  {
    float linv[8], pmv[8];
    for (int v = 0; v < 8; ++v) {
      const int row = rb * 16 + 8 * half + v;
      linv[v] = 1.f / fmaxf(lrow[row], 1e-30f);
      pmv[v]  = (float)mask_p[b * TP + i0 + row];
    }
    for (int tt = 0; tt < 8; ++tt) {
      const int c0 = (ch * 8 + tt) * 16;
      for (int v = 0; v < 8; ++v) {
        const int row = rb * 16 + 8 * half + v;
        G[((size_t)b * TP + i0 + row) * GW + DD + c0 + r] =
            Oa[tt][v] * linv[v] * pmv[v];
      }
    }
  }
  {
    const int row = tid >> 2, seg = tid & 3;
    const float pm = (float)mask_p[b * TP + i0 + row];
    const float* hrow = H + ((size_t)b * TP + i0 + row) * DD + seg * 64;
    float* grow = G + ((size_t)b * TP + i0 + row) * GW + seg * 64;
    for (int c = 0; c < 64; c += 4) {
      float4 hv = *(const float4*)(hrow + c);
      hv.x *= pm; hv.y *= pm; hv.z *= pm; hv.w *= pm;
      *(float4*)(grow + c) = hv;
    }
  }
}

// ---------------------------------------------------------------------------
// Kernel 3: per-batch softmax of Mrow over i (masked by pm) -> h_att[b][c]
// ---------------------------------------------------------------------------
__global__ __launch_bounds__(256)
void reduce_kernel(const float* __restrict__ H, const int* __restrict__ mask_p,
                   const float* __restrict__ Mrow, float* __restrict__ hatt)
{
  __shared__ float red[256];
  __shared__ float smax, ssum;
  const int b = blockIdx.x, tid = threadIdx.x;

  float lm = -3.0e38f;
  for (int i = tid; i < TP; i += 256) lm = fmaxf(lm, Mrow[b * TP + i]);
  red[tid] = lm; __syncthreads();
  for (int s = 128; s > 0; s >>= 1) {
    if (tid < s) red[tid] = fmaxf(red[tid], red[tid + s]);
    __syncthreads();
  }
  if (tid == 0) smax = red[0];
  __syncthreads();
  const float mx = smax;

  float ls = 0.f;
  for (int i = tid; i < TP; i += 256) {
    const float pm = (float)mask_p[b * TP + i];
    ls += pm * __expf(pm * (Mrow[b * TP + i] - mx));
  }
  red[tid] = ls; __syncthreads();
  for (int s = 128; s > 0; s >>= 1) {
    if (tid < s) red[tid] += red[tid + s];
    __syncthreads();
  }
  if (tid == 0) ssum = red[0];
  __syncthreads();
  const float zinv = 1.f / fmaxf(ssum, 1e-30f);

  const int c = tid;                 // 256 threads = 256 channels
  float acc = 0.f;
  for (int i = 0; i < TP; ++i) {
    const float pm = (float)mask_p[b * TP + i];
    const float wgt = pm * __expf(pm * (Mrow[b * TP + i] - mx)) * zinv;
    acc += wgt * pm * H[((size_t)b * TP + i) * DD + c];
  }
  hatt[b * DD + c] = acc;
}

// ---------------------------------------------------------------------------
// Kernel 4: fill Hm*U_att and Hm*h_att slices of G
// ---------------------------------------------------------------------------
__global__ __launch_bounds__(256)
void epilogue_kernel(const float* __restrict__ hatt, float* __restrict__ G)
{
  const size_t rowid = blockIdx.x;        // 0 .. BB*TP-1
  const int b = (int)(rowid / TP);
  const int c = threadIdx.x;
  float* g = G + rowid * GW;
  const float hm = g[c];
  const float ua = g[DD + c];
  g[2 * DD + c] = hm * ua;
  g[3 * DD + c] = hm * hatt[b * DD + c];
}

// ---------------------------------------------------------------------------
extern "C" void kernel_launch(void* const* d_in, const int* in_sizes, int n_in,
                              void* d_out, int out_size, void* d_ws, size_t ws_size,
                              hipStream_t stream)
{
  const float* H      = (const float*)d_in[0];
  const float* U      = (const float*)d_in[1];
  const float* w      = (const float*)d_in[2];
  const int*   mask_p = (const int*)d_in[3];
  const int*   mask_q = (const int*)d_in[4];
  float* G = (float*)d_out;

  char* ws = (char*)d_ws;
  unsigned short* Ubf  = (unsigned short*)(ws);                       // 4 MB
  unsigned short* Utbf = (unsigned short*)(ws + (4u << 20));          // 4 MB
  float* cU   = (float*)(ws + (8u << 20));                            // 32 KB
  float* Mrow = (float*)(ws + (8u << 20) + (64u << 10));              // 256 KB
  float* hatt = (float*)(ws + (8u << 20) + (512u << 10));             // 16 KB

  prep_u_kernel<<<dim3((BB * TQ) / 256), dim3(256), 0, stream>>>(
      U, w, mask_q, Ubf, Utbf, cU);

  attn_kernel<<<dim3(TP / ROWS, BB), dim3(256), 0, stream>>>(
      H, w, mask_p, mask_q, Ubf, Utbf, cU, Mrow, G);

  reduce_kernel<<<dim3(BB), dim3(256), 0, stream>>>(H, mask_p, Mrow, hatt);

  epilogue_kernel<<<dim3(BB * TP), dim3(256), 0, stream>>>(hatt, G);
}
